// RHN_13786845020503
// MI455X (gfx1250) — matrix operand order repaired and checked by reading, not compiled, not run on hardware
//
#include <hip/hip_runtime.h>
#include <math.h>

#define RHN_LAYERS 2
#define RHN_DEPTH  4
#define RHN_H      1024
#define RHN_IN     1024
#define RHN_SEQ    512
#define RHN_B      64
#define RHN_2H     2048

typedef __attribute__((ext_vector_type(16))) __bf16          v16bf;
typedef __attribute__((ext_vector_type(8)))  float           v8f;
typedef __attribute__((ext_vector_type(16))) unsigned short  us16;
typedef __attribute__((ext_vector_type(8)))  unsigned short  us8;

// AMDGCN fast-math builtins exist only in the device pass; host pass still
// parses __device__ bodies, so give it a host-compilable fallback.
#if defined(__HIP_DEVICE_COMPILE__) && __has_builtin(__builtin_amdgcn_exp2f)
#define FAST_EXP2(x) __builtin_amdgcn_exp2f(x)
#else
#define FAST_EXP2(x) exp2f(x)
#endif
#if defined(__HIP_DEVICE_COMPILE__) && __has_builtin(__builtin_amdgcn_rcpf)
#define FAST_RCP(x) __builtin_amdgcn_rcpf(x)
#else
#define FAST_RCP(x) (1.0f / (x))
#endif

__device__ __forceinline__ unsigned short f2bf(float f) {
    unsigned int u = __float_as_uint(f);
    unsigned int r = u + 0x7FFFu + ((u >> 16) & 1u);   // round-to-nearest-even
    return (unsigned short)(r >> 16);
}

// Branch-free gates: one v_exp_f32 + one v_rcp_f32 each, clamped so the
// ratio forms never hit inf/inf.
__device__ __forceinline__ float fast_sigmoid(float x) {
    x = fminf(fmaxf(x, -30.0f), 30.0f);
    float e = FAST_EXP2(-1.4426950408889634f * x);     // exp(-x)
    return FAST_RCP(1.0f + e);
}
__device__ __forceinline__ float fast_tanh(float x) {
    x = fminf(fmaxf(x, -15.0f), 15.0f);
    float e = FAST_EXP2(2.8853900817779268f * x);      // exp(2x)
    return (e - 1.0f) * FAST_RCP(e + 1.0f);
}

// ---------------- prepass kernels ----------------

__global__ void rhn_cvt_x(const float* __restrict__ x, unsigned short* __restrict__ xbf, int n) {
    int i = blockIdx.x * blockDim.x + threadIdx.x;
    int stride = gridDim.x * blockDim.x;
    for (; i < n; i += stride) xbf[i] = f2bf(x[i]);
}

// W0[l][n][k] (n:2048, k:2048): k<1024 -> w_h[l][0][k][n], else w_in[l][k-1024][n]
__global__ void rhn_build_w0(const float* __restrict__ w_in, const float* __restrict__ w_h,
                             unsigned short* __restrict__ W0) {
    const long total = 2L * 2048 * 2048;
    long i = (long)blockIdx.x * blockDim.x + threadIdx.x;
    long stride = (long)gridDim.x * blockDim.x;
    for (; i < total; i += stride) {
        long l = i >> 22;
        long rem = i & ((1L << 22) - 1);
        long n = rem >> 11;
        long k = rem & 2047;
        float v = (k < 1024)
            ? w_h[(((size_t)l * RHN_DEPTH + 0) * 1024 + k) * 2048 + n]
            : w_in[((size_t)l * 1024 + (k - 1024)) * 2048 + n];
        W0[i] = f2bf(v);
    }
}

// Wd[l][d-1][n][k] (n:2048, k:1024) = w_h[l][d][k][n], d=1..3
__global__ void rhn_build_wd(const float* __restrict__ w_h, unsigned short* __restrict__ Wd) {
    const long total = 2L * 3 * 2048 * 1024;
    long i = (long)blockIdx.x * blockDim.x + threadIdx.x;
    long stride = (long)gridDim.x * blockDim.x;
    for (; i < total; i += stride) {
        long ld  = i / (2048L * 1024);
        long rem = i % (2048L * 1024);
        long l = ld / 3;
        long d = (ld % 3) + 1;
        long n = rem >> 10;
        long k = rem & 1023;
        Wd[i] = f2bf(w_h[(((size_t)l * RHN_DEPTH + d) * 1024 + k) * 2048 + n]);
    }
}

// Init bf16 state ping-pong buffers from fp32 state, zero Inp1, reset barrier.
__global__ void rhn_init_state(const float* __restrict__ st,
                               unsigned short* __restrict__ Sbf,
                               unsigned short* __restrict__ Inp1,
                               int* __restrict__ bar) {
    const int SN = RHN_B * RHN_H;   // 65536
    int i = blockIdx.x * blockDim.x + threadIdx.x;
    int stride = gridDim.x * blockDim.x;
    for (int j = i; j < 2 * SN; j += stride) {
        int l = j / SN, e = j % SN;
        unsigned short b = f2bf(st[(size_t)l * SN + e]);
        Sbf[(size_t)(l * 2 + 0) * SN + e] = b;
        Sbf[(size_t)(l * 2 + 1) * SN + e] = b;
    }
    for (int j = i; j < SN; j += stride) Inp1[j] = 0;
    if (i == 0) { bar[0] = 0; bar[1] = 0; }
}

// ---------------- device-wide barrier ----------------

__device__ __forceinline__ void rhn_grid_sync(int* bar, int nblocks) {
    __threadfence();
    __syncthreads();
    if (threadIdx.x == 0) {
        int gen = __hip_atomic_load(&bar[1], __ATOMIC_RELAXED, __HIP_MEMORY_SCOPE_AGENT);
        int arrived = __hip_atomic_fetch_add(&bar[0], 1, __ATOMIC_ACQ_REL, __HIP_MEMORY_SCOPE_AGENT);
        if (arrived == nblocks - 1) {
            __hip_atomic_store(&bar[0], 0, __ATOMIC_RELAXED, __HIP_MEMORY_SCOPE_AGENT);
            __hip_atomic_fetch_add(&bar[1], 1, __ATOMIC_RELEASE, __HIP_MEMORY_SCOPE_AGENT);
        } else {
            while (__hip_atomic_load(&bar[1], __ATOMIC_ACQUIRE, __HIP_MEMORY_SCOPE_AGENT) == gen) {
                __builtin_amdgcn_s_sleep(2);
            }
        }
    }
    __syncthreads();
}

// ---------------- K-loop: register double-buffered WMMA pair ----------------
// aS/aI have the per-lane A offset (row*H + ko) folded in; wH/wG have
// (col*K + kB) folded in. CAT selects the K=2048 concat-A path (depth 0).

template<bool CAT>
__device__ __forceinline__ void rhn_gemm_pair(
    const unsigned short* __restrict__ aS,
    const unsigned short* __restrict__ aI,
    const unsigned short* __restrict__ wH,
    const unsigned short* __restrict__ wG,
    const int K, v8f& accH, v8f& accG)
{
    us8  a0c, a1c;
    us16 bhc, bgc;
    {
        const unsigned short* ap = aS;     // k = 0 is always the state half
        a0c = *(const us8*)(ap);
        a1c = *(const us8*)(ap + 8);
        bhc = *(const us16*)(wH);
        bgc = *(const us16*)(wG);
    }
    #pragma unroll 2
    for (int k = 0; k < K; k += 32) {
        const int kn = (k + 32 < K) ? (k + 32) : k;          // tail: dead reload
        const unsigned short* apn = (CAT && kn >= RHN_H) ? (aI + (kn - RHN_H)) : (aS + kn);
        us8  a0n = *(const us8*)(apn);
        us8  a1n = *(const us8*)(apn + 8);
        us16 bhn = *(const us16*)(wH + kn);
        us16 bgn = *(const us16*)(wG + kn);
        __builtin_prefetch(wH + k + 256, 0, 1);              // L2->near stream prefetch
        __builtin_prefetch(wG + k + 256, 0, 1);
        union { us8 h[2]; v16bf v; } a; a.h[0] = a0c; a.h[1] = a1c;
        union { us16 u; v16bf v; } bh, bg; bh.u = bhc; bg.u = bgc;
        accH = __builtin_amdgcn_wmma_f32_16x16x32_bf16(
            false, a.v, false, bh.v, (short)0, accH, false, false);
        accG = __builtin_amdgcn_wmma_f32_16x16x32_bf16(
            false, a.v, false, bg.v, (short)0, accG, false, false);
        a0c = a0n; a1c = a1n; bhc = bhn; bgc = bgn;
    }
}

// ---------------- persistent RHN kernel ----------------
// grid = 32 blocks x 256 threads = 256 waves; wave p owns output tile
// (tm = p&3, tn = p>>2): batch rows tm*16..+15, HIDDEN cols tn*16..+15
// (h half) paired with cols +1024 (g half).

__global__ __launch_bounds__(256, 1) void rhn_persistent(
    const unsigned short* __restrict__ Xbf,   // [SEQ][64][1024] bf16
    unsigned short* __restrict__ Sbf,         // [2 layers][2 bufs][64*1024] bf16
    unsigned short* __restrict__ Inp1,        // [64*1024] bf16 (layer-1 input)
    const unsigned short* __restrict__ W0,    // [2][2048 n][2048 k] bf16
    const unsigned short* __restrict__ Wd,    // [2][3][2048 n][1024 k] bf16
    const float* __restrict__ bias,           // [2][4][2048] fp32
    const float* __restrict__ st,             // [2][64][1024] fp32 initial state
    float* __restrict__ out,                  // SEQ*64*1024 then 2*64*1024
    int* __restrict__ bar, int nblocks)
{
    const int SN   = RHN_B * RHN_H;           // 65536
    const int lane = threadIdx.x & 31;
    const int wave = threadIdx.x >> 5;
    const int p    = blockIdx.x * (blockDim.x >> 5) + wave;  // 0..255
    const int tm   = p & 3;
    const int tn   = p >> 2;                  // 0..63
    const int arow = tm * 16 + (lane & 15);   // A-fragment row
    const int ko   = (lane >> 4) * 16;        // A K-offset per lane half
    const int kB   = (lane >> 4) * 16;        // B K-offset per lane half
    const int colH = tn * 16 + (lane & 15);   // B/C column
    const int hi8  = (lane >> 4) * 8;         // C-tile row offset

    const size_t aoff = (size_t)arow * RHN_H + ko;
    const unsigned short* inp1_a = Inp1 + aoff;

    // fp32 carry state + biases live in registers (l,d loops fully unrolled
    // below => constant indices, no scratch)
    float sreg[2][8];
    float biasH[2][4], biasG[2][4];
    #pragma unroll
    for (int l = 0; l < 2; ++l) {
        #pragma unroll
        for (int r = 0; r < 8; ++r)
            sreg[l][r] = st[(size_t)l * SN + (size_t)(tm * 16 + hi8 + r) * RHN_H + colH];
        #pragma unroll
        for (int d = 0; d < 4; ++d) {
            biasH[l][d] = bias[(size_t)(l * RHN_DEPTH + d) * RHN_2H + colH];
            biasG[l][d] = bias[(size_t)(l * RHN_DEPTH + d) * RHN_2H + RHN_H + colH];
        }
    }

    #pragma unroll 1
    for (int t = 0; t < RHN_SEQ; ++t) {
        const unsigned short* xt_a = Xbf + (size_t)t * SN + aoff;
        #pragma unroll
        for (int l = 0; l < RHN_LAYERS; ++l) {
            const unsigned short* aI = (l == 0) ? xt_a : inp1_a;
            #pragma unroll
            for (int d = 0; d < RHN_DEPTH; ++d) {
                const unsigned short* aS = Sbf + (size_t)(l * 2 + (d & 1)) * SN + aoff;
                v8f accH = {};
                v8f accG = {};
                if (d == 0) {
                    const unsigned short* W = W0 + (size_t)l * RHN_2H * 2048;
                    rhn_gemm_pair<true>(aS, aI,
                        W + (size_t)colH * 2048 + kB,
                        W + (size_t)(colH + RHN_H) * 2048 + kB,
                        2048, accH, accG);
                } else {
                    const unsigned short* W = Wd + (size_t)(l * 3 + (d - 1)) * RHN_2H * 1024;
                    rhn_gemm_pair<false>(aS, nullptr,
                        W + (size_t)colH * 1024 + kB,
                        W + (size_t)(colH + RHN_H) * 1024 + kB,
                        1024, accH, accG);
                }

                unsigned short* So = Sbf + (size_t)(l * 2 + ((d + 1) & 1)) * SN;
                #pragma unroll
                for (int r = 0; r < 8; ++r) {
                    const int orow = tm * 16 + hi8 + r;
                    float hv = fast_tanh(accH[r] + biasH[l][d]);
                    float gv = fast_sigmoid(accG[r] + biasG[l][d]);
                    float sn = hv * gv + sreg[l][r] * (1.0f - gv);
                    sreg[l][r] = sn;
                    unsigned short sb = f2bf(sn);
                    So[(size_t)orow * RHN_H + colH] = sb;
                    if (l == 0 && d == RHN_DEPTH - 1)
                        Inp1[(size_t)orow * RHN_H + colH] = sb;
                    if (l == 1 && d == RHN_DEPTH - 1)
                        out[(size_t)t * SN + (size_t)orow * RHN_H + colH] = sn;
                }
                rhn_grid_sync(bar, nblocks);
            }
        }
    }

    // final states: appended after the [SEQ,B,H] output
    #pragma unroll
    for (int l = 0; l < 2; ++l)
        #pragma unroll
        for (int r = 0; r < 8; ++r) {
            const int orow = tm * 16 + hi8 + r;
            out[(size_t)RHN_SEQ * SN + (size_t)l * SN + (size_t)orow * RHN_H + colH] = sreg[l][r];
        }
}

// ---------------- host launcher ----------------

extern "C" void kernel_launch(void* const* d_in, const int* in_sizes, int n_in,
                              void* d_out, int out_size, void* d_ws, size_t ws_size,
                              hipStream_t stream) {
    const float* x    = (const float*)d_in[0];   // [512,64,1024]
    const float* st   = (const float*)d_in[1];   // [2,64,1024]
    const float* w_in = (const float*)d_in[2];   // [2,1024,2048]
    const float* w_h  = (const float*)d_in[3];   // [2,4,1024,2048]
    const float* b_h  = (const float*)d_in[4];   // [2,4,2048]
    float* out = (float*)d_out;

    char* ws = (char*)d_ws;
    const size_t SN = (size_t)RHN_B * RHN_H;                 // 65536 elems
    const size_t BAR_OFF  = 0;                               // 256 B
    const size_t XBF_OFF  = 256;
    const size_t XBF_N    = (size_t)RHN_SEQ * SN;            // 33,554,432
    const size_t SBF_OFF  = XBF_OFF + XBF_N * 2;
    const size_t SBF_N    = 4 * SN;                          // 2 layers x 2 bufs
    const size_t INP1_OFF = SBF_OFF + SBF_N * 2;
    const size_t INP1_N   = SN;
    const size_t W0_OFF   = INP1_OFF + INP1_N * 2;
    const size_t W0_N     = 2ull * 2048 * 2048;
    const size_t WD_OFF   = W0_OFF + W0_N * 2;
    const size_t WD_N     = 2ull * 3 * 2048 * 1024;
    const size_t TOTAL    = WD_OFF + WD_N * 2;               // ~105 MB
    if (ws_size < TOTAL) return;

    int*            bar  = (int*)(ws + BAR_OFF);
    unsigned short* Xbf  = (unsigned short*)(ws + XBF_OFF);
    unsigned short* Sbf  = (unsigned short*)(ws + SBF_OFF);
    unsigned short* Inp1 = (unsigned short*)(ws + INP1_OFF);
    unsigned short* W0   = (unsigned short*)(ws + W0_OFF);
    unsigned short* Wd   = (unsigned short*)(ws + WD_OFF);

    rhn_cvt_x   <<<4096, 256, 0, stream>>>(x, Xbf, (int)XBF_N);
    rhn_build_w0<<<4096, 256, 0, stream>>>(w_in, w_h, W0);
    rhn_build_wd<<<4096, 256, 0, stream>>>(w_h, Wd);
    rhn_init_state<<<512, 256, 0, stream>>>(st, Sbf, Inp1, bar);

    const int NBLOCKS = 32;   // 32 blocks x 8 waves = 256 waves = 256 tile-pairs
    rhn_persistent<<<NBLOCKS, 256, 0, stream>>>(Xbf, Sbf, Inp1, W0, Wd, b_h, st, out,
                                                bar, NBLOCKS);
}
